// LUCE_42451456754143
// MI455X (gfx1250) — matrix operand order repaired and checked by reading, not compiled, not run on hardware
//
#include <hip/hip_runtime.h>
#include <hip/hip_bf16.h>

// ---------------- problem constants (match reference) ----------------
#define NV   2048        // num nodes
#define ETN  4           // edge types
#define CCH  2           // channels
#define NEE  32768       // edges per type
#define WIN  64
#define WOUT 64
#define HID  128
#define NN   ((size_t)NV * (size_t)NV)

typedef unsigned short u16;
typedef __attribute__((ext_vector_type(16))) __bf16 v16bf;
typedef __attribute__((ext_vector_type(8)))  float  v8f;

// gfx1250 async global->LDS copy (ASYNCcnt-tracked), with sync fallback
#ifndef USE_ASYNC_LDS
#if __has_builtin(__builtin_amdgcn_global_load_async_to_lds_b128)
#define USE_ASYNC_LDS 1
#else
#define USE_ASYNC_LDS 0
#endif
#endif

#if USE_ASYNC_LDS
typedef int v4i_t __attribute__((vector_size(16)));
typedef __attribute__((address_space(1))) v4i_t* g_v4i_p;   // global (AS1)
typedef __attribute__((address_space(3))) v4i_t* l_v4i_p;   // LDS (AS3)

__device__ __forceinline__ void async_cp16(const u16* gsrc, u16* ldst) {
    // flat LDS addresses carry the LDS byte offset in the low 32 bits
    __builtin_amdgcn_global_load_async_to_lds_b128(
        (g_v4i_p)(size_t)gsrc,
        (l_v4i_p)(unsigned)(size_t)ldst,
        0, 0);
}
__device__ __forceinline__ void async_wait0() {
#if __has_builtin(__builtin_amdgcn_s_wait_asynccnt)
    __builtin_amdgcn_s_wait_asynccnt(0);
#else
    asm volatile("s_wait_asynccnt 0" ::: "memory");
#endif
}
#else
__device__ __forceinline__ void async_wait0() {}
#endif

// float -> bf16 (round to nearest even, good enough for well-scaled data)
__device__ __forceinline__ u16 f2bf(float f) {
    unsigned int u = __float_as_uint(f);
    unsigned int r = (u + 0x7FFFu + ((u >> 16) & 1u)) >> 16;
    return (u16)r;
}

__device__ __forceinline__ float sigm(float x) { return 1.0f / (1.0f + expf(-x)); }

// ---------------- 1. zero fill ----------------
__global__ void zero_f32(float* p, size_t n) {
    size_t i = (size_t)blockIdx.x * blockDim.x + threadIdx.x;
    size_t stride = (size_t)gridDim.x * blockDim.x;
    for (; i < n; i += stride) p[i] = 0.0f;
}

// ---------------- 2. scatter-add edges into dense A [ET,N,N] ----------------
__global__ void scatter_edges(const int* __restrict__ ei, const float* __restrict__ ev,
                              float* __restrict__ A) {
    int t = blockIdx.x * blockDim.x + threadIdx.x;
    if (t >= ETN * NEE) return;
    int e = t / NEE, k = t - e * NEE;
    int r = ei[(size_t)e * 2 * NEE + k];
    int c = ei[(size_t)e * 2 * NEE + NEE + k];
    atomicAdd(&A[(size_t)e * NN + (size_t)r * NV + c], ev[(size_t)e * NEE + k]);
}

// ---------------- 3a. softmax over edge types for the 6 (layer,channel) rows ---
__global__ void softmax_gtw(const float* __restrict__ gtw, float* __restrict__ s) {
    int t = threadIdx.x;                       // rows: 3 layers * 2 channels
    if (t >= 6) return;
    float w[ETN], m = -1e30f, sum = 0.f;
    for (int e = 0; e < ETN; ++e) { w[e] = gtw[t * ETN + e]; m = fmaxf(m, w[e]); }
    for (int e = 0; e < ETN; ++e) { w[e] = expf(w[e] - m); sum += w[e]; }
    for (int e = 0; e < ETN; ++e) s[t * ETN + e] = w[e] / sum;
}

// ---------------- 3b. fused mixing: M_l[c] = sum_e s[l,c,e] * A[e] (bf16 out) --
__global__ void mix_adj(const float* __restrict__ A, const float* __restrict__ s,
                        u16* __restrict__ M0, u16* __restrict__ M1, u16* __restrict__ M2) {
    size_t i = (size_t)blockIdx.x * blockDim.x + threadIdx.x;
    size_t stride = (size_t)gridDim.x * blockDim.x;
    for (; i < NN; i += stride) {
        float a0 = A[i], a1 = A[NN + i], a2 = A[2 * NN + i], a3 = A[3 * NN + i];
        u16* outs[3] = {M0, M1, M2};
        #pragma unroll
        for (int l = 0; l < 3; ++l)
            #pragma unroll
            for (int c = 0; c < CCH; ++c) {
                const float* w = &s[(l * CCH + c) * ETN];
                float v = w[0] * a0 + w[1] * a1 + w[2] * a2 + w[3] * a3;
                outs[l][(size_t)c * NN + i] = f2bf(v);
            }
    }
}

// ---------------- 4. bf16 WMMA GEMM: C[c] = A[c] @ B[c], f32 out ----------------
// macro tile 128x128, TILE_K=32, 256 threads = 8 waves in a 4(M) x 2(N) grid,
// each wave computes 32x64 = 2x4 WMMA 16x16 tiles. Double-buffered LDS with
// async global->LDS for A, register-staged transpose for B, prefetch of t+2.
#define LDT 40   // padded LDS row stride in u16 (80B: keeps 16B alignment for b128)

__global__ __launch_bounds__(256)
void wmma_gemm_bf16(const u16* __restrict__ Abase, const u16* __restrict__ Bbase,
                    float* __restrict__ Cbase) {
    const int ch = blockIdx.z;
    const u16* Ag = Abase + (size_t)ch * NN;
    const u16* Bg = Bbase + (size_t)ch * NN;
    float*     Cg = Cbase + (size_t)ch * NN;

    __shared__ __align__(16) u16 As[2][128 * LDT];
    __shared__ __align__(16) u16 Bs[2][128 * LDT];   // stored transposed: Bs[b][n][k]

    const int tid  = threadIdx.x;
    const int lane = tid & 31, wave = tid >> 5;
    const int waveM = wave & 3, waveN = wave >> 2;
    const int rowBase = blockIdx.y * 128;
    const int colBase = blockIdx.x * 128;
    const int mm = lane & 15, hi = lane >> 4;

    // per-thread staging roles
    const int am = tid >> 1, akk = (tid & 1) << 4;      // A: 128 rows x 32 k
    const int bkk = tid >> 3, bn0 = (tid & 7) << 4;     // B: 32 k x 128 n

    const int T = NV / 32;
    v8f acc[2][4] = {};

    // ---- prologue: stage tile 0 into buffer 0 ----
    {
        const u16* gA = Ag + (size_t)(rowBase + am) * NV + akk;
        u16* lA = &As[0][am * LDT + akk];
#if USE_ASYNC_LDS
        async_cp16(gA, lA);
        async_cp16(gA + 8, lA + 8);
#else
        const uint4* src = (const uint4*)gA;
        *(uint4*)lA       = src[0];
        *(uint4*)(lA + 8) = src[1];
#endif
        const uint4* srcB = (const uint4*)(Bg + (size_t)bkk * NV + colBase + bn0);
        union { uint4 q[2]; u16 u[16]; } tb;
        tb.q[0] = srcB[0]; tb.q[1] = srcB[1];
        #pragma unroll
        for (int j = 0; j < 16; ++j) Bs[0][(bn0 + j) * LDT + bkk] = tb.u[j];
    }
    async_wait0();
    __syncthreads();

    for (int i = 0; i < T; ++i) {
        const int cur = i & 1, nxt = cur ^ 1;
        const bool more = (i + 1 < T);

        // ---- kick off tile i+1 (overlaps with WMMAs below) ----
        uint4 aq0 = {}, aq1 = {}, bq0 = {}, bq1 = {};
        if (more) {
            const int k0n = (i + 1) * 32;
            const u16* gA = Ag + (size_t)(rowBase + am) * NV + k0n + akk;
#if USE_ASYNC_LDS
            u16* lA = &As[nxt][am * LDT + akk];
            async_cp16(gA, lA);
            async_cp16(gA + 8, lA + 8);
#else
            const uint4* srcA = (const uint4*)gA;
            aq0 = srcA[0]; aq1 = srcA[1];
#endif
            const uint4* srcB = (const uint4*)(Bg + (size_t)(k0n + bkk) * NV + colBase + bn0);
            bq0 = srcB[0]; bq1 = srcB[1];
            if (i + 2 < T) {   // L2 prefetch of tile i+2 -> global_prefetch_b8
                const int k0p = (i + 2) * 32;
                __builtin_prefetch(Ag + (size_t)(rowBase + am) * NV + k0p + akk, 0, 1);
                __builtin_prefetch(Bg + (size_t)(k0p + bkk) * NV + colBase + bn0, 0, 1);
            }
        }

        // ---- compute on buffer `cur` ----
        // A fragments: lane<16 holds K0-15; lane>=16 K16-31 (matches B)
        v16bf aF[2], bF[4];
        #pragma unroll
        for (int mf = 0; mf < 2; ++mf) {
            const u16* r = &As[cur][(waveM * 32 + mf * 16 + mm) * LDT];
            int k1 = hi ? 16 : 0, k2 = k1 + 8;
            union { uint4 q[2]; v16bf v; } u;
            u.q[0] = *(const uint4*)(r + k1);
            u.q[1] = *(const uint4*)(r + k2);
            aF[mf] = u.v;
        }
        // B fragments: lane n = lane%16, k = (lane/16)*16 + e  (Bs is [n][k])
        #pragma unroll
        for (int nf = 0; nf < 4; ++nf) {
            const u16* r = &Bs[cur][(waveN * 64 + nf * 16 + mm) * LDT + (hi << 4)];
            union { uint4 q[2]; v16bf v; } u;
            u.q[0] = *(const uint4*)(r);
            u.q[1] = *(const uint4*)(r + 8);
            bF[nf] = u.v;
        }
        #pragma unroll
        for (int mf = 0; mf < 2; ++mf)
            #pragma unroll
            for (int nf = 0; nf < 4; ++nf)
                acc[mf][nf] = __builtin_amdgcn_wmma_f32_16x16x32_bf16(
                    false, aF[mf], false, bF[nf], (short)0, acc[mf][nf], false, false);

        // ---- finish staging tile i+1 into buffer `nxt` ----
        if (more) {
#if !USE_ASYNC_LDS
            u16* lA = &As[nxt][am * LDT + akk];
            *(uint4*)lA       = aq0;
            *(uint4*)(lA + 8) = aq1;
#endif
            union { uint4 q[2]; u16 u[16]; } tb;
            tb.q[0] = bq0; tb.q[1] = bq1;
            #pragma unroll
            for (int j = 0; j < 16; ++j) Bs[nxt][(bn0 + j) * LDT + bkk] = tb.u[j];
            async_wait0();
            __syncthreads();
        }
    }

    // C layout: VGPR v -> row v + 8*hi, col = mm
    #pragma unroll
    for (int mf = 0; mf < 2; ++mf)
        #pragma unroll
        for (int nf = 0; nf < 4; ++nf) {
            int row0 = rowBase + waveM * 32 + mf * 16 + hi * 8;
            int col  = colBase + waveN * 64 + nf * 16 + mm;
            #pragma unroll
            for (int v = 0; v < 8; ++v)
                Cg[(size_t)(row0 + v) * NV + col] = acc[mf][nf][v];
        }
}

// ---------------- 5. row-wise D^-1 normalize; emit bf16 or f32 ----------------
__global__ void norm_rows(const float* __restrict__ in, u16* __restrict__ outBf,
                          float* __restrict__ outF, int writeBf) {
    const int r = blockIdx.x;                     // over C*N rows
    const float* row = in + (size_t)r * NV;
    __shared__ float red[256];
    float s = 0.f;
    for (int i = threadIdx.x; i < NV; i += 256) s += row[i];
    red[threadIdx.x] = s;
    __syncthreads();
    for (int off = 128; off > 0; off >>= 1) {
        if (threadIdx.x < off) red[threadIdx.x] += red[threadIdx.x + off];
        __syncthreads();
    }
    float total = red[0];
    float dinv = (total == 0.0f) ? 0.0f : 1.0f / total;
    for (int i = threadIdx.x; i < NV; i += 256) {
        float v = row[i] * dinv;
        if (writeBf) outBf[(size_t)r * NV + i] = f2bf(v);
        else         outF[(size_t)r * NV + i] = v;
    }
}

// ---------------- 6. XW = X @ W_gcn ----------------
__global__ void xw_kernel(const float* __restrict__ X, const float* __restrict__ Wg,
                          float* __restrict__ XW) {
    int idx = blockIdx.x * blockDim.x + threadIdx.x;   // N*64
    if (idx >= NV * WOUT) return;
    int n = idx >> 6, f = idx & 63;
    float acc = 0.f;
    #pragma unroll 4
    for (int k = 0; k < WIN; ++k) acc += X[n * WIN + k] * Wg[k * WOUT + f];
    XW[idx] = acc;
}

// ---------------- 7. Xc = relu(H2 @ XW + b); write as X_ [N, C*64] ----------------
__global__ void einsum_relu(const float* __restrict__ H2, const float* __restrict__ XW,
                            const float* __restrict__ b, float* __restrict__ Xo) {
    const int n = blockIdx.x, c = blockIdx.y, f = threadIdx.x;   // 64 threads/block
    const float* hrow = H2 + ((size_t)c * NV + n) * NV;
    float acc = 0.f;
    #pragma unroll 4
    for (int m = 0; m < NV; ++m) acc += hrow[m] * XW[m * WOUT + f];
    acc += b[f];
    Xo[(size_t)n * (CCH * WOUT) + c * WOUT + f] = fmaxf(acc, 0.0f);
}

// ---------------- 8. G_in = X_ @ W_ih^T + b_ih + b_hh  [N, 4*HID] ----------------
__global__ void gin_kernel(const float* __restrict__ Xo, const float* __restrict__ Wih,
                           const float* __restrict__ bih, const float* __restrict__ bhh,
                           float* __restrict__ G) {
    int idx = blockIdx.x * blockDim.x + threadIdx.x;   // N * 512
    if (idx >= NV * 4 * HID) return;
    int n = idx >> 9, r = idx & 511;
    float acc = bih[r] + bhh[r];
    const float* xr = &Xo[(size_t)n * (CCH * WOUT)];
    const float* wr = &Wih[(size_t)r * (CCH * WOUT)];
    #pragma unroll 8
    for (int k = 0; k < CCH * WOUT; ++k) acc += xr[k] * wr[k];
    G[idx] = acc;
}

// ---------------- 9. sequential LSTM, single persistent workgroup ----------------
// W_hh (256KB) lives transposed in LDS (CDNA5 WGP has 320KB LDS). 512 threads,
// one gate row each; 128 conflict-free LDS FMAs per step on the serial path.
__global__ __launch_bounds__(512)
void lstm_kernel(const float* __restrict__ G, const float* __restrict__ Whh,
                 float* __restrict__ hs) {
    extern __shared__ float sm[];
    float* whh  = sm;                       // [128][512] transposed: whh[k*512+r]
    float* hbuf = sm + 128 * 512;           // [128]
    float* cbuf = hbuf + 128;               // [128]
    float* gbuf = cbuf + 128;               // [512]
    const int tid = threadIdx.x;

    // load W_hh transposed (coalesced global reads)
    for (int i = tid; i < 4 * HID * HID; i += 512) {
        int r = i >> 7, k = i & 127;
        whh[k * 512 + r] = Whh[i];
    }
    if (tid < HID) { hbuf[tid] = 0.f; cbuf[tid] = 0.f; }
    __syncthreads();

    for (int n = 0; n < NV; ++n) {
        float acc = G[(size_t)n * 512 + tid];
        #pragma unroll 8
        for (int k = 0; k < HID; ++k) acc += whh[k * 512 + tid] * hbuf[k];
        gbuf[tid] = acc;
        __syncthreads();
        if (tid < HID) {
            float gi = gbuf[tid], gf = gbuf[HID + tid];
            float gg = gbuf[2 * HID + tid], go = gbuf[3 * HID + tid];
            float c = sigm(gf) * cbuf[tid] + sigm(gi) * tanhf(gg);
            float h = sigm(go) * tanhf(c);
            cbuf[tid] = c; hbuf[tid] = h;
            hs[(size_t)n * HID + tid] = h;
        }
        __syncthreads();
    }
}

// ---------------- 10. y = leaky_relu(hs @ W_lin^T + b, 0.2) ----------------
__global__ void head_kernel(const float* __restrict__ hs, const float* __restrict__ Wl,
                            const float* __restrict__ bl, float* __restrict__ out) {
    int n = blockIdx.x * blockDim.x + threadIdx.x;
    if (n >= NV) return;
    float acc = bl[0];
    #pragma unroll 8
    for (int k = 0; k < HID; ++k) acc += hs[(size_t)n * HID + k] * Wl[k];
    out[n] = (acc >= 0.f) ? acc : 0.2f * acc;
}

// ---------------- launch ----------------
extern "C" void kernel_launch(void* const* d_in, const int* in_sizes, int n_in,
                              void* d_out, int out_size, void* d_ws, size_t ws_size,
                              hipStream_t stream) {
    const int*   edge_index = (const int*)  d_in[0];
    const float* edge_value = (const float*)d_in[1];
    const float* X          = (const float*)d_in[2];
    // d_in[3] = target (unused in forward)
    const float* gt_w       = (const float*)d_in[4];
    const float* W_gcn      = (const float*)d_in[5];
    const float* b_gcn      = (const float*)d_in[6];
    const float* W_ih       = (const float*)d_in[7];
    const float* W_hh       = (const float*)d_in[8];
    const float* b_ih       = (const float*)d_in[9];
    const float* b_hh       = (const float*)d_in[10];
    const float* W_lin      = (const float*)d_in[11];
    const float* b_lin      = (const float*)d_in[12];
    float* out = (float*)d_out;

    // carve workspace
    size_t off = 0;
    char* base = (char*)d_ws;
    auto carve = [&](size_t bytes) -> char* {
        char* p = base + off;
        off += (bytes + 255) & ~(size_t)255;
        return p;
    };
    float* A    = (float*)carve(ETN * NN * sizeof(float));        // 64 MB
    u16*   M0b  = (u16*)  carve(CCH * NN * sizeof(u16));          // 16 MB
    u16*   M1b  = (u16*)  carve(CCH * NN * sizeof(u16));
    u16*   M2b  = (u16*)  carve(CCH * NN * sizeof(u16));
    float* Hf   = (float*)carve(CCH * NN * sizeof(float));        // 32 MB
    u16*   Hb   = (u16*)  carve(CCH * NN * sizeof(u16));          // 16 MB
    float* H2   = (float*)carve(CCH * NN * sizeof(float));        // 32 MB
    float* sfx  = (float*)carve(3 * CCH * ETN * sizeof(float));
    float* XW   = (float*)carve((size_t)NV * WOUT * sizeof(float));
    float* Xo   = (float*)carve((size_t)NV * CCH * WOUT * sizeof(float));
    float* Gin  = (float*)carve((size_t)NV * 4 * HID * sizeof(float));
    float* hsB  = (float*)carve((size_t)NV * HID * sizeof(float));
    (void)ws_size; (void)in_sizes; (void)n_in; (void)out_size;

    // 1-2: dense adjacencies
    zero_f32<<<4096, 256, 0, stream>>>(A, ETN * NN);
    scatter_edges<<<(ETN * NEE + 255) / 256, 256, 0, stream>>>(edge_index, edge_value, A);

    // 3: softmax weights + fused bf16 mixes
    softmax_gtw<<<1, 32, 0, stream>>>(gt_w, sfx);
    mix_adj<<<4096, 256, 0, stream>>>(A, sfx, M0b, M1b, M2b);

    // 4-5: H = normalize(mix0 @ mix1), then H2 = normalize(H @ mix2)
    dim3 gg(NV / 128, NV / 128, CCH);
    wmma_gemm_bf16<<<gg, 256, 0, stream>>>(M0b, M1b, Hf);
    norm_rows<<<CCH * NV, 256, 0, stream>>>(Hf, Hb, nullptr, 1);
    wmma_gemm_bf16<<<gg, 256, 0, stream>>>(Hb, M2b, H2);
    norm_rows<<<CCH * NV, 256, 0, stream>>>(H2, nullptr, H2, 0);   // f32 in-place

    // 6-7: shared GCN
    xw_kernel<<<(NV * WOUT + 255) / 256, 256, 0, stream>>>(X, W_gcn, XW);
    dim3 ge(NV, CCH);
    einsum_relu<<<ge, WOUT, 0, stream>>>(H2, XW, b_gcn, Xo);

    // 8-9: LSTM (input GEMM hoisted out of the serial loop)
    gin_kernel<<<(NV * 4 * HID + 255) / 256, 256, 0, stream>>>(Xo, W_ih, b_ih, b_hh, Gin);
    size_t lstm_smem = (size_t)(128 * 512 + 128 + 128 + 512) * sizeof(float);
    lstm_kernel<<<1, 512, lstm_smem, stream>>>(Gin, W_hh, hsB);

    // 10: linear head + LeakyReLU(0.2)
    head_kernel<<<(NV + 255) / 256, 256, 0, stream>>>(hsB, W_lin, b_lin, out);
}
